// MultiHeadSelfAttention_42898133353001
// MI455X (gfx1250) — matrix operand order repaired and checked by reading, not compiled, not run on hardware
//
#include <hip/hip_runtime.h>
#include <stdint.h>

typedef __bf16 bf16_t;
typedef __attribute__((ext_vector_type(16))) __bf16 bf16x16;
typedef __attribute__((ext_vector_type(8)))  __bf16 bf16x8;
typedef __attribute__((ext_vector_type(2)))  __bf16 bf16x2;
typedef __attribute__((ext_vector_type(8)))  float  f32x8;

static constexpr int S  = 4096;
static constexpr int H  = 768;
static constexpr int NH = 12;
static constexpr int HD = 64;
static constexpr int H3 = 2304;

static __device__ __forceinline__ bf16x16 make_frag(bf16x8 lo, bf16x8 hi) {
  return __builtin_shufflevector(lo, hi, 0,1,2,3,4,5,6,7,8,9,10,11,12,13,14,15);
}
static __device__ __forceinline__ bf16x8 cvt8(f32x8 x) {
  bf16x8 r;
#pragma unroll
  for (int i = 0; i < 8; ++i) r[i] = (bf16_t)x[i];
  return r;
}
static __device__ __forceinline__ f32x8 zero8() {
  f32x8 z;
#pragma unroll
  for (int i = 0; i < 8; ++i) z[i] = 0.f;
  return z;
}
static __device__ __forceinline__ f32x8 wmma_bf16(bf16x16 a, bf16x16 b, f32x8 c) {
  return __builtin_amdgcn_wmma_f32_16x16x32_bf16(false, a, false, b, (short)0, c, false, false);
}

// ---------------------------------------------------------------- mask bias
__global__ void mask_bias_kernel(const unsigned char* __restrict__ m,
                                 float* __restrict__ b) {
  int i = blockIdx.x * blockDim.x + threadIdx.x;
  if (i < S) b[i] = m[i] ? -1e30f : 0.0f;
}

// ------------------------------------------------- fused QKV projection GEMM
// C[4096,2304] = X[4096,768] @ W[768,2304] + b ; scatter into q/k/vT bf16.
// 128x64 tile per block; each wave owns 32 rows (2 M-tiles).
// Q/K/V routing is uniform per block (n0 multiple of 64; boundaries 768/1536
// are multiples of 64) -> scalar branch, straight-line stores.
__global__ __launch_bounds__(128) void qkv_proj_kernel(
    const float* __restrict__ X, const float* __restrict__ W,
    const float* __restrict__ Bq,
    bf16_t* __restrict__ qh, bf16_t* __restrict__ kh, bf16_t* __restrict__ vT) {
  __shared__ __align__(16) bf16_t WT[64][40];  // [n][k], 80B rows (16B mult)
  const int tid = threadIdx.x, lane = tid & 31, wave = tid >> 5;
  const int lane15 = lane & 15, hi = lane >> 4, rowoff = hi * 8;
  const int m0 = blockIdx.x * 128, n0 = blockIdx.y * 64;

  f32x8 acc[2][4];
#pragma unroll
  for (int mt = 0; mt < 2; ++mt)
#pragma unroll
    for (int nt = 0; nt < 4; ++nt) acc[mt][nt] = zero8();

  const float* Xr[2];
  Xr[0] = X + (size_t)(m0 + wave * 32 + lane15) * H;
  Xr[1] = Xr[0] + (size_t)16 * H;

  for (int k0 = 0; k0 < H; k0 += 32) {
    const int kb = k0 + hi * 8;
    // global A loads issued before the barrier to overlap with staging
    f32x8 fa[2][2];
#pragma unroll
    for (int mt = 0; mt < 2; ++mt) {
      fa[mt][0] = *(const f32x8*)(Xr[mt] + kb);
      fa[mt][1] = *(const f32x8*)(Xr[mt] + kb + 16);
    }
    __syncthreads();
#pragma unroll
    for (int t = 0; t < 16; ++t) {
      int idx = t * 128 + tid;
      int kl = idx >> 6, nl = idx & 63;
      WT[nl][kl] = (bf16_t)W[(size_t)(k0 + kl) * H3 + n0 + nl];
    }
    __syncthreads();

    bf16x16 afrag[2];
#pragma unroll
    for (int mt = 0; mt < 2; ++mt)
      afrag[mt] = make_frag(cvt8(fa[mt][0]), cvt8(fa[mt][1]));
    bf16x16 bfrag[4];
#pragma unroll
    for (int nt = 0; nt < 4; ++nt) {
      const bf16_t* bp = &WT[nt * 16 + lane15][hi * 8];
      bfrag[nt] = make_frag(*(const bf16x8*)bp, *(const bf16x8*)(bp + 16));
    }
#pragma unroll
    for (int nt = 0; nt < 4; ++nt)
#pragma unroll
      for (int mt = 0; mt < 2; ++mt)
        acc[mt][nt] = wmma_bf16(afrag[mt], bfrag[nt], acc[mt][nt]);
  }

  // ---- epilogue: scalar routing on n0 (uniform across the block)
  const int rbase = m0 + wave * 32 + rowoff;
  float bb[4];
#pragma unroll
  for (int nt = 0; nt < 4; ++nt) bb[nt] = Bq[n0 + nt * 16 + lane15];

  if (n0 < H) {                          // ---- Q (pre-scaled by sqrt(64)=8)
    const int h = n0 >> 6;               // n0 % 64 == 0 -> head uniform
    bf16_t* dst = qh + (size_t)h * S * HD;
#pragma unroll
    for (int mt = 0; mt < 2; ++mt)
#pragma unroll
      for (int nt = 0; nt < 4; ++nt) {
        const int d = nt * 16 + lane15;
#pragma unroll
        for (int r = 0; r < 8; ++r) {
          const int row = rbase + mt * 16 + r;
          dst[(size_t)row * HD + d] = (bf16_t)((acc[mt][nt][r] + bb[nt]) * 8.0f);
        }
      }
  } else if (n0 < 2 * H) {               // ---- K
    const int h = (n0 - H) >> 6;
    bf16_t* dst = kh + (size_t)h * S * HD;
#pragma unroll
    for (int mt = 0; mt < 2; ++mt)
#pragma unroll
      for (int nt = 0; nt < 4; ++nt) {
        const int d = nt * 16 + lane15;
#pragma unroll
        for (int r = 0; r < 8; ++r) {
          const int row = rbase + mt * 16 + r;
          dst[(size_t)row * HD + d] = (bf16_t)(acc[mt][nt][r] + bb[nt]);
        }
      }
  } else {                               // ---- V transposed: [head][d][seq]
    const int h = (n0 - 2 * H) >> 6;
    bf16_t* dst = vT + (size_t)h * HD * S;
#pragma unroll
    for (int mt = 0; mt < 2; ++mt)
#pragma unroll
      for (int nt = 0; nt < 4; ++nt) {
        const int d = nt * 16 + lane15;
#pragma unroll
        for (int r = 0; r < 8; ++r) {
          const int row = rbase + mt * 16 + r;
          dst[(size_t)d * S + row] = (bf16_t)(acc[mt][nt][r] + bb[nt]);
        }
      }
  }
}

// ------------------------------------------------------- flash attention
// grid (S/128, NH); 4 waves; wave owns 32 query rows; key blocks of 32.
__global__ __launch_bounds__(128) void flash_attn_kernel(
    const bf16_t* __restrict__ qh, const bf16_t* __restrict__ kh,
    const bf16_t* __restrict__ vT, const float* __restrict__ mb,
    bf16_t* __restrict__ attnOut) {
  __shared__ __align__(16) float  Sld[4][32][36];  // 144B rows
  __shared__ __align__(16) bf16_t Pld[4][32][32];
  __shared__ float rowsc[4][32];

  const int tid = threadIdx.x, lane = tid & 31, w = tid >> 5;
  const int lane15 = lane & 15, hi = lane >> 4, rowoff = hi * 8;
  const int head = blockIdx.y;
  const int q0 = blockIdx.x * 128 + w * 32;

  const bf16_t* Q = qh + (size_t)head * S * HD;
  const bf16_t* K = kh + (size_t)head * S * HD;
  const bf16_t* V = vT + (size_t)head * HD * S;

  // Q A-fragments: 2 M-tiles x 2 K-chunks
  bf16x16 aq[2][2];
#pragma unroll
  for (int mt = 0; mt < 2; ++mt) {
    const bf16_t* qr = Q + (size_t)(q0 + mt * 16 + lane15) * HD + hi * 8;
    aq[mt][0] = make_frag(*(const bf16x8*)(qr),      *(const bf16x8*)(qr + 16));
    aq[mt][1] = make_frag(*(const bf16x8*)(qr + 32), *(const bf16x8*)(qr + 48));
  }

  f32x8 o[2][4];
#pragma unroll
  for (int mt = 0; mt < 2; ++mt)
#pragma unroll
    for (int dt = 0; dt < 4; ++dt) o[mt][dt] = zero8();
  float mrun = -1e30f, lrun = 0.0f;   // lane L owns row q0+L

  for (int j0 = 0; j0 < S; j0 += 32) {
    // prefetch next key block (global_prefetch_b8; no counter cost)
    if (j0 + 32 < S) {
      __builtin_prefetch(K + (size_t)(j0 + 32 + lane) * HD, 0, 0);
      __builtin_prefetch(V + (size_t)lane * S + j0 + 32, 0, 0);
      __builtin_prefetch(V + (size_t)(lane + 32) * S + j0 + 32, 0, 0);
    }

    // ---- K fragments for this block (2 N-tiles x 2 K-chunks)
    bf16x16 kf[2][2];
#pragma unroll
    for (int nt = 0; nt < 2; ++nt) {
      const bf16_t* kp = K + (size_t)(j0 + nt * 16 + lane15) * HD + hi * 8;
      kf[nt][0] = make_frag(*(const bf16x8*)kp,        *(const bf16x8*)(kp + 16));
      kf[nt][1] = make_frag(*(const bf16x8*)(kp + 32), *(const bf16x8*)(kp + 48));
    }
    const float cb[2] = { mb[j0 + lane15], mb[j0 + 16 + lane15] };

    // ---- S = (Q*8) @ K^T + bias -> per-wave LDS tile (C layout)
#pragma unroll
    for (int mt = 0; mt < 2; ++mt) {
#pragma unroll
      for (int nt = 0; nt < 2; ++nt) {
        f32x8 s = zero8();
        s = wmma_bf16(aq[mt][0], kf[nt][0], s);
        s = wmma_bf16(aq[mt][1], kf[nt][1], s);
#pragma unroll
        for (int r = 0; r < 8; ++r)
          Sld[w][mt * 16 + r + rowoff][nt * 16 + lane15] = s[r] + cb[nt];
      }
    }
    __syncthreads();

    // ---- online softmax: lane L owns row L of the wave's 32-row tile
    {
      const float4* srow = (const float4*)(&Sld[w][lane][0]);
      float4 sv[8];
#pragma unroll
      for (int t = 0; t < 8; ++t) sv[t] = srow[t];
      float bmax = -1e30f;
#pragma unroll
      for (int t = 0; t < 8; ++t) {
        bmax = fmaxf(bmax, sv[t].x); bmax = fmaxf(bmax, sv[t].y);
        bmax = fmaxf(bmax, sv[t].z); bmax = fmaxf(bmax, sv[t].w);
      }
      float mnew = fmaxf(mrun, bmax);
      float sc = __expf(mrun - mnew);
      float ss = 0.f;
#pragma unroll
      for (int t = 0; t < 8; ++t) {
        float p0 = __expf(sv[t].x - mnew);
        float p1 = __expf(sv[t].y - mnew);
        float p2 = __expf(sv[t].z - mnew);
        float p3 = __expf(sv[t].w - mnew);
        ss += (p0 + p1) + (p2 + p3);
        bf16x2 a = { (bf16_t)p0, (bf16_t)p1 };
        bf16x2 b = { (bf16_t)p2, (bf16_t)p3 };
        *(bf16x2*)(&Pld[w][lane][t * 4])     = a;
        *(bf16x2*)(&Pld[w][lane][t * 4 + 2]) = b;
      }
      lrun = lrun * sc + ss;
      mrun = mnew;
      rowsc[w][lane] = sc;
    }
    __syncthreads();

    // ---- rescale O, then O += P @ V
#pragma unroll
    for (int mt = 0; mt < 2; ++mt)
#pragma unroll
      for (int r = 0; r < 8; ++r) {
        float sc = rowsc[w][mt * 16 + r + rowoff];
        o[mt][0][r] *= sc; o[mt][1][r] *= sc;
        o[mt][2][r] *= sc; o[mt][3][r] *= sc;
      }
    bf16x16 pa[2];
#pragma unroll
    for (int mt = 0; mt < 2; ++mt) {
      const bf16_t* pp = &Pld[w][mt * 16 + lane15][hi * 8];
      pa[mt] = make_frag(*(const bf16x8*)pp, *(const bf16x8*)(pp + 16));
    }
#pragma unroll
    for (int dt = 0; dt < 4; ++dt) {
      const bf16_t* vp = V + (size_t)(dt * 16 + lane15) * S + j0 + hi * 8;
      bf16x16 vb = make_frag(*(const bf16x8*)vp, *(const bf16x8*)(vp + 16));
#pragma unroll
      for (int mt = 0; mt < 2; ++mt)
        o[mt][dt] = wmma_bf16(pa[mt], vb, o[mt][dt]);
    }
    // per-wave LDS regions + in-order DS per wave: 2 barriers/iter suffice
  }

  rowsc[w][lane] = 1.0f / lrun;
  __syncthreads();
#pragma unroll
  for (int mt = 0; mt < 2; ++mt)
#pragma unroll
    for (int r = 0; r < 8; ++r) {
      const float inv = rowsc[w][mt * 16 + r + rowoff];
      const int row = q0 + mt * 16 + r + rowoff;
      bf16_t* op = attnOut + (size_t)row * H + head * HD;
#pragma unroll
      for (int dt = 0; dt < 4; ++dt)
        op[dt * 16 + lane15] = (bf16_t)(o[mt][dt][r] * inv);
    }
}

// ------------------------------------------------------ output projection
// 128x64 tile per block; wave owns 32 rows; A already bf16 row-major.
__global__ __launch_bounds__(128) void out_proj_kernel(
    const bf16_t* __restrict__ A, const float* __restrict__ W,
    const float* __restrict__ Bo, float* __restrict__ out) {
  __shared__ __align__(16) bf16_t WT[64][40];
  const int tid = threadIdx.x, lane = tid & 31, wave = tid >> 5;
  const int lane15 = lane & 15, hi = lane >> 4, rowoff = hi * 8;
  const int m0 = blockIdx.x * 128, n0 = blockIdx.y * 64;

  f32x8 acc[2][4];
#pragma unroll
  for (int mt = 0; mt < 2; ++mt)
#pragma unroll
    for (int nt = 0; nt < 4; ++nt) acc[mt][nt] = zero8();

  const bf16_t* Ar[2];
  Ar[0] = A + (size_t)(m0 + wave * 32 + lane15) * H;
  Ar[1] = Ar[0] + (size_t)16 * H;

  for (int k0 = 0; k0 < H; k0 += 32) {
    const int kb = k0 + hi * 8;
    bf16x8 la[2][2];
#pragma unroll
    for (int mt = 0; mt < 2; ++mt) {
      la[mt][0] = *(const bf16x8*)(Ar[mt] + kb);
      la[mt][1] = *(const bf16x8*)(Ar[mt] + kb + 16);
    }
    __syncthreads();
#pragma unroll
    for (int t = 0; t < 16; ++t) {
      int idx = t * 128 + tid;
      int kl = idx >> 6, nl = idx & 63;
      WT[nl][kl] = (bf16_t)W[(size_t)(k0 + kl) * H + n0 + nl];
    }
    __syncthreads();

    bf16x16 afrag[2];
#pragma unroll
    for (int mt = 0; mt < 2; ++mt) afrag[mt] = make_frag(la[mt][0], la[mt][1]);
    bf16x16 bfrag[4];
#pragma unroll
    for (int nt = 0; nt < 4; ++nt) {
      const bf16_t* bp = &WT[nt * 16 + lane15][hi * 8];
      bfrag[nt] = make_frag(*(const bf16x8*)bp, *(const bf16x8*)(bp + 16));
    }
#pragma unroll
    for (int nt = 0; nt < 4; ++nt)
#pragma unroll
      for (int mt = 0; mt < 2; ++mt)
        acc[mt][nt] = wmma_bf16(afrag[mt], bfrag[nt], acc[mt][nt]);
  }

#pragma unroll
  for (int mt = 0; mt < 2; ++mt)
#pragma unroll
    for (int nt = 0; nt < 4; ++nt) {
      const int col = n0 + nt * 16 + lane15;
      const float bb = Bo[col];
#pragma unroll
      for (int r = 0; r < 8; ++r) {
        const int row = m0 + wave * 32 + mt * 16 + r + rowoff;
        out[(size_t)row * H + col] = acc[mt][nt][r] + bb;
      }
    }
}

// ---------------------------------------------------------------- launcher
extern "C" void kernel_launch(void* const* d_in, const int* in_sizes, int n_in,
                              void* d_out, int out_size, void* d_ws, size_t ws_size,
                              hipStream_t stream) {
  const float* qkv           = (const float*)d_in[0];
  const unsigned char* mask  = (const unsigned char*)d_in[1];
  const float* w_qkv         = (const float*)d_in[2];
  const float* b_qkv         = (const float*)d_in[3];
  const float* w_out         = (const float*)d_in[4];
  const float* b_out         = (const float*)d_in[5];
  float* out = (float*)d_out;

  char* ws = (char*)d_ws;
  const size_t szQ = (size_t)NH * S * HD * sizeof(bf16_t);  // 6,291,456 B
  bf16_t* qh   = (bf16_t*)(ws);
  bf16_t* kh   = (bf16_t*)(ws + szQ);
  bf16_t* vT   = (bf16_t*)(ws + 2 * szQ);
  bf16_t* attn = (bf16_t*)(ws + 3 * szQ);
  float*  mbias = (float*)(ws + 4 * szQ);

  mask_bias_kernel<<<dim3((S + 255) / 256), dim3(256), 0, stream>>>(mask, mbias);
  qkv_proj_kernel<<<dim3(S / 128, H3 / 64), dim3(128), 0, stream>>>(
      qkv, w_qkv, b_qkv, qh, kh, vT);
  flash_attn_kernel<<<dim3(S / 128, NH), dim3(128), 0, stream>>>(
      qh, kh, vT, mbias, attn);
  out_proj_kernel<<<dim3(S / 128, H / 64), dim3(128), 0, stream>>>(
      attn, w_out, b_out, out);
}
